// Restormer_Decoder_61057255080409
// MI455X (gfx1250) — matrix operand order repaired and checked by reading, not compiled, not run on hardware
//
#include <hip/hip_runtime.h>
#include <hip/hip_bf16.h>

// MI455X (gfx1250, wave32). All dense 1x1-conv GEMMs run on
// v_wmma_f32_16x16x32_f16 with f16-staged activations in LDS and f16
// weights pre-converted once into workspace. Depthwise 3x3 convs are fused
// into their consumers; bias/residual/eca-gate fused into WMMA prologue.
// All hot-loop addressing uses uniform SGPR base + 32-bit offsets so the
// backend emits saddr+voffset global ops instead of per-lane 64-bit adds.

#define HW   65536   // 256*256
#define BATCH 4

typedef __attribute__((ext_vector_type(16))) _Float16 v16h;
typedef __attribute__((ext_vector_type(8)))  float    v8f;

__device__ __forceinline__ float sigmoidf_(float x) { return 1.f / (1.f + expf(-x)); }
__device__ __forceinline__ float geluf_(float x) { return 0.5f * x * (1.f + erff(x * 0.70710678118f)); }

// ---------------------------------------------------------------------------
// Generic 1x1-conv GEMM: Y[b, ocOff+oc, p] = sum_ic W[oc,ic] * X[b, icOff+ic, p]
// Block = 256 threads = 8 waves; each wave -> 16oc x 16pix WMMA tile.
// Activation panel (IC x 32 pixels) staged in LDS as f16.
// ---------------------------------------------------------------------------
template<int IC, bool BIAS, bool RES, bool GATE, bool DUAL>
__global__ __launch_bounds__(256) void gemm1x1(
    const float* __restrict__ X, int ICT, int icOff,
    const _Float16* __restrict__ W16,
    const float* __restrict__ bias,
    const float* __restrict__ gatev,      // [b*ICT + ic] per-channel gate (eca)
    const float* __restrict__ res,        // residual, output layout
    float* __restrict__ Y, float* __restrict__ Y2,
    int OC, int OCT, int ocOff)
{
  __shared__ _Float16 xs[IC * 32];
  const int b  = blockIdx.y;
  const int p0 = blockIdx.x * 32;
  // uniform (scalar) base pointers; all per-lane offsets are 32-bit
  const float* __restrict__ Xb = X + (size_t)b * ICT * HW + (size_t)icOff * HW;
  float* __restrict__ Yb  = Y + (size_t)b * OCT * HW + (size_t)ocOff * HW;
  float* __restrict__ Y2b = DUAL ? (Y2 + (size_t)b * OCT * HW + (size_t)ocOff * HW) : nullptr;
  const float* __restrict__ Rb = RES ? (res + (size_t)b * OCT * HW + (size_t)ocOff * HW) : nullptr;

  // stage IC x 32 activation panel (f32 -> f16)
  for (int idx = threadIdx.x; idx < IC * 32; idx += 256) {
    int ic = idx >> 5, px = idx & 31;
    float v = Xb[(unsigned)(ic * HW + p0 + px)];
    if (GATE) v *= gatev[b * ICT + icOff + ic];
    xs[idx] = (_Float16)v;
  }
  __syncthreads();

  const int wave   = threadIdx.x >> 5;
  const int lane   = threadIdx.x & 31;
  const int ocSub  = (wave & 3) * 16;   // 4 oc sub-tiles
  const int pixSub = (wave >> 2) * 16;  // 2 pixel sub-tiles
  const int col    = lane & 15;
  const int hi     = lane >> 4;         // lane half
  const unsigned p = (unsigned)(p0 + pixSub + col);
  const int pxl    = pixSub + col;

  for (int ocb = 0; ocb < OC; ocb += 64) {
    const int ocBase = ocb + ocSub;
    // accumulator preload: bias + residual (WMMA computes D = A*B + C)
    v8f acc;
#pragma unroll
    for (int v = 0; v < 8; ++v) {
      int oc = ocBase + v + hi * 8;
      float a = 0.f;
      if (BIAS) a += bias[oc];
      if (RES)  a += Rb[(unsigned)(oc * HW) + p];
      acc[v] = a;
    }
    for (int k0 = 0; k0 < IC; k0 += 32) {
      // A fragment (weights, 16x32 f16) per documented VGPR layout
      v16h af;
      const _Float16* wrow = W16 + (size_t)(ocBase + col) * IC + k0;
      const int kb = hi * 8;
#pragma unroll
      for (int j = 0; j < 4; ++j) {
        af[2 * j]     = wrow[kb + 2 * j];
        af[2 * j + 1] = wrow[kb + 2 * j + 1];
      }
#pragma unroll
      for (int j = 4; j < 8; ++j) {
        af[2 * j]     = wrow[16 + kb + 2 * (j - 4)];
        af[2 * j + 1] = wrow[16 + kb + 2 * (j - 4) + 1];
      }
      // B fragment (activations, 32x16 f16) from LDS
      v16h bf;
      const int khalf = hi * 8;
#pragma unroll
      for (int j = 0; j < 8; ++j) {
        bf[2 * j]     = xs[(k0 + khalf + 2 * j + (j >> 2) * 8) * 32 + pxl];
        bf[2 * j + 1] = xs[(k0 + khalf + 2 * j + 1 + (j >> 2) * 8) * 32 + pxl];
      }
      acc = __builtin_amdgcn_wmma_f32_16x16x32_f16(false, af, false, bf,
                                                   (short)0, acc, false, false);
    }
    // epilogue: scatter f32 per C/D layout (32-bit offsets from scalar base)
#pragma unroll
    for (int v = 0; v < 8; ++v) {
      int oc = ocBase + v + hi * 8;
      unsigned o = (unsigned)(oc * HW) + p;
      Yb[o] = acc[v];
      if (DUAL) Y2b[o] = acc[v];
    }
  }
}

// ---------------------------------------------------------------------------
// small helpers
// ---------------------------------------------------------------------------
__global__ void k_cvt(const float* __restrict__ s, _Float16* __restrict__ d, int n) {
  int i = blockIdx.x * 256 + threadIdx.x;
  if (i < n) d[i] = (_Float16)s[i];
}

__global__ void k_zero(float* p, int n) {
  int i = blockIdx.x * 256 + threadIdx.x;
  if (i < n) p[i] = 0.f;
}

// per-(b,c) mean over HW
__global__ void k_chan_mean(const float* __restrict__ X, float* __restrict__ out, int C) {
  int c = blockIdx.x, b = blockIdx.y;
  const float* p = X + (size_t)b * C * HW + (size_t)c * HW;
  float s = 0.f;
  for (int i = threadIdx.x; i < HW; i += 256) s += p[i];
  __shared__ float red[256];
  red[threadIdx.x] = s; __syncthreads();
  for (int st = 128; st > 0; st >>= 1) {
    if (threadIdx.x < st) red[threadIdx.x] += red[threadIdx.x + st];
    __syncthreads();
  }
  if (threadIdx.x == 0) out[b * C + c] = red[0] * (1.f / (float)HW);
}

__global__ void k_eca(const float* __restrict__ means, const float* __restrict__ ecaw,
                      float* __restrict__ gate) {
  int i = blockIdx.x * 256 + threadIdx.x;
  if (i < BATCH * 128) gate[i] = sigmoidf_(means[i] * ecaw[i & 127]);
}

// ---------------------------------------------------------------------------
// CIIM grouped depthwise conv (6 in-ch/group over concat(fuse, la, ha)),
// gating, concat(h*low, l*high), channel shuffle (g=32).
// ---------------------------------------------------------------------------
__global__ void k_ciim_dw(const float* __restrict__ fuse,    // [b][128][HW]
                          const float* __restrict__ meansLH, // [b][256] (lf,hf means)
                          const float* __restrict__ low, const float* __restrict__ high,
                          const float* __restrict__ wdw,     // [64][6][9]
                          const float* __restrict__ bdw,     // [64]
                          float* __restrict__ out)           // [b][128][HW] shuffled
{
  int b = blockIdx.z, g = blockIdx.y;
  int p = blockIdx.x * 256 + threadIdx.x;
  int y = p >> 8, x = p & 255;
  const float* fb = fuse + (size_t)b * 128 * HW;
  float acc = bdw[g];
  for (int j = 0; j < 6; ++j) {
    int ch = 6 * g + j;
    const float* w = wdw + (size_t)(g * 6 + j) * 9;
    if (ch < 128) {
      for (int dy = -1; dy <= 1; ++dy)
        for (int dx = -1; dx <= 1; ++dx) {
          int yy = y + dy, xx = x + dx;
          if (yy < 0 || yy > 255 || xx < 0 || xx > 255) continue;
          acc += w[(dy + 1) * 3 + (dx + 1)] * fb[(unsigned)(ch * HW + yy * 256 + xx)];
        }
    } else {
      float mv = (ch < 256) ? meansLH[b * 256 + (ch - 128)]
                            : meansLH[b * 256 + 128 + (ch - 256)];
      for (int dy = -1; dy <= 1; ++dy)
        for (int dx = -1; dx <= 1; ++dx) {
          int yy = y + dy, xx = x + dx;
          if (yy < 0 || yy > 255 || xx < 0 || xx > 255) continue;
          acc += w[(dy + 1) * 3 + (dx + 1)] * mv;  // spatially-constant channel
        }
    }
  }
  float hmap = sigmoidf_(geluf_(acc));
  float lmap = sigmoidf_(fmaxf(acc, 0.f));
  float o1 = hmap * low [(size_t)b * 64 * HW + (unsigned)(g * HW + p)];
  float o2 = lmap * high[(size_t)b * 64 * HW + (unsigned)(g * HW + p)];
  int c1 = g, c2 = g + 64;  // shuffle: c -> (c%4)*32 + c/4  (groups = 32)
  int d1 = (c1 & 3) * 32 + (c1 >> 2);
  int d2 = (c2 & 3) * 32 + (c2 >> 2);
  float* ob = out + (size_t)b * 128 * HW;
  ob[(unsigned)(d1 * HW + p)] = o1;
  ob[(unsigned)(d2 * HW + p)] = o2;
}

// ---------------------------------------------------------------------------
// channel LayerNorm (per pixel over 64 channels)
// ---------------------------------------------------------------------------
__global__ void k_ln(const float* __restrict__ X, const float* __restrict__ w,
                     const float* __restrict__ bv, float* __restrict__ Y)
{
  int b = blockIdx.y;
  unsigned p = blockIdx.x * 256 + threadIdx.x;
  const float* xb = X + (size_t)b * 64 * HW;
  float* yb = Y + (size_t)b * 64 * HW;
  float v[64], mu = 0.f;
#pragma unroll
  for (int c = 0; c < 64; ++c) { v[c] = xb[(unsigned)(c * HW) + p]; mu += v[c]; }
  mu *= (1.f / 64.f);
  float var = 0.f;
#pragma unroll
  for (int c = 0; c < 64; ++c) { float d = v[c] - mu; var += d * d; }
  var *= (1.f / 64.f);
  float inv = rsqrtf(var + 1e-5f);
#pragma unroll
  for (int c = 0; c < 64; ++c)
    yb[(unsigned)(c * HW) + p] = (v[c] - mu) * inv * w[c] + bv[c];
}

// ---------------------------------------------------------------------------
// attention Gram: per (b,h) accumulate q-row/k-row sumsq + 8x8 q.k^T over
// all 65536 pixels; depthwise 3x3 fused on the fly.
// gram layout per (b,h): [0:8)=ssq, [8:16)=ssk, [16:80)=G
// ---------------------------------------------------------------------------
__global__ void k_gram(const float* __restrict__ qkv, const float* __restrict__ wdw,
                       float* __restrict__ gram)
{
  int b = blockIdx.z, h = blockIdx.y, chunk = blockIdx.x;
  __shared__ float acc[80];
  for (int i = threadIdx.x; i < 80; i += 256) acc[i] = 0.f;
  __syncthreads();
  const float* base = qkv + (size_t)b * 192 * HW;
  float ssq[8] = {0}, ssk[8] = {0}, G[64] = {0};
  for (int t = 0; t < 4; ++t) {
    int p = chunk * 1024 + t * 256 + threadIdx.x;
    int y = p >> 8, x = p & 255;
    float qv[8], kv[8];
#pragma unroll
    for (int c = 0; c < 8; ++c) {
      int qc = h * 8 + c, kc = 64 + h * 8 + c;
      float sq = 0.f, sk = 0.f;
      for (int dy = -1; dy <= 1; ++dy)
        for (int dx = -1; dx <= 1; ++dx) {
          int yy = y + dy, xx = x + dx;
          if (yy < 0 || yy > 255 || xx < 0 || xx > 255) continue;
          int pp = yy * 256 + xx, wi = (dy + 1) * 3 + (dx + 1);
          sq += wdw[qc * 9 + wi] * base[(unsigned)(qc * HW + pp)];
          sk += wdw[kc * 9 + wi] * base[(unsigned)(kc * HW + pp)];
        }
      qv[c] = sq; kv[c] = sk;
      ssq[c] += sq * sq; ssk[c] += sk * sk;
    }
#pragma unroll
    for (int c = 0; c < 8; ++c)
#pragma unroll
      for (int d = 0; d < 8; ++d) G[c * 8 + d] += qv[c] * kv[d];
  }
  for (int c = 0; c < 8; ++c) { atomicAdd(&acc[c], ssq[c]); atomicAdd(&acc[8 + c], ssk[c]); }
  for (int i = 0; i < 64; ++i) atomicAdd(&acc[16 + i], G[i]);
  __syncthreads();
  float* g = gram + (size_t)(b * 8 + h) * 80;
  for (int i = threadIdx.x; i < 80; i += 256) atomicAdd(&g[i], acc[i]);
}

__global__ void k_softmax(const float* __restrict__ gram, const float* __restrict__ temp,
                          float* __restrict__ attn)
{
  int h = blockIdx.x, b = blockIdx.y;
  const float* g = gram + (size_t)(b * 8 + h) * 80;
  __shared__ float s[64];
  int c = threadIdx.x >> 3, d = threadIdx.x & 7;
  float nq = fmaxf(sqrtf(g[c]), 1e-12f);
  float nk = fmaxf(sqrtf(g[8 + d]), 1e-12f);
  float sv = g[16 + c * 8 + d] / (nq * nk) * temp[h];
  s[threadIdx.x] = sv;
  __syncthreads();
  float m = -1e30f;
  for (int j = 0; j < 8; ++j) m = fmaxf(m, s[c * 8 + j]);
  float sum = 0.f;
  for (int j = 0; j < 8; ++j) sum += expf(s[c * 8 + j] - m);
  attn[(size_t)(b * 8 + h) * 64 + c * 8 + d] = expf(sv - m) / sum;
}

// out[b, h*8+c, p] = sum_d attn[c,d] * dw3x3(v)[d, p]   (v = qkv ch 128..191)
__global__ void k_attn_apply(const float* __restrict__ qkv, const float* __restrict__ wdw,
                             const float* __restrict__ attn, float* __restrict__ out)
{
  int b = blockIdx.y;
  int p = blockIdx.x * 256 + threadIdx.x;
  int y = p >> 8, x = p & 255;
  const float* base = qkv + (size_t)b * 192 * HW;
  float* ob = out + (size_t)b * 64 * HW;
  for (int h = 0; h < 8; ++h) {
    float dv[8];
#pragma unroll
    for (int c = 0; c < 8; ++c) {
      int vc = 128 + h * 8 + c;
      float s = 0.f;
      for (int dy = -1; dy <= 1; ++dy)
        for (int dx = -1; dx <= 1; ++dx) {
          int yy = y + dy, xx = x + dx;
          if (yy < 0 || yy > 255 || xx < 0 || xx > 255) continue;
          s += wdw[vc * 9 + (dy + 1) * 3 + (dx + 1)] * base[(unsigned)(vc * HW + yy * 256 + xx)];
        }
      dv[c] = s;
    }
    const float* a = attn + (size_t)(b * 8 + h) * 64;
#pragma unroll
    for (int c = 0; c < 8; ++c) {
      float o = 0.f;
#pragma unroll
      for (int d = 0; d < 8; ++d) o += a[c * 8 + d] * dv[d];
      ob[(unsigned)((h * 8 + c) * HW + p)] = o;
    }
  }
}

// FFN: fused depthwise 3x3 + gelu gating: out[c] = gelu(dw(y,c)) * dw(y,c+128)
__global__ void k_ffn_gate(const float* __restrict__ yy, const float* __restrict__ wdw,
                           float* __restrict__ out)
{
  int b = blockIdx.y;
  int p = blockIdx.x * 256 + threadIdx.x;
  int y = p >> 8, x = p & 255;
  const float* base = yy + (size_t)b * 256 * HW;
  float* ob = out + (size_t)b * 128 * HW;
  for (int c = 0; c < 128; ++c) {
    float d1 = 0.f, d2 = 0.f;
    for (int dy = -1; dy <= 1; ++dy)
      for (int dx = -1; dx <= 1; ++dx) {
        int yyp = y + dy, xx = x + dx;
        if (yyp < 0 || yyp > 255 || xx < 0 || xx > 255) continue;
        int pp = yyp * 256 + xx, wi = (dy + 1) * 3 + (dx + 1);
        d1 += wdw[c * 9 + wi] * base[(unsigned)(c * HW + pp)];
        d2 += wdw[(c + 128) * 9 + wi] * base[(unsigned)((c + 128) * HW + pp)];
      }
    ob[(unsigned)(c * HW + p)] = geluf_(d1) * d2;
  }
}

// head conv 3x3 64->32 + LeakyReLU(0.01); blockIdx.y selects 8-oc group
__global__ void k_head1(const float* __restrict__ X, const float* __restrict__ w,
                        float* __restrict__ out)
{
  int b = blockIdx.z, oq = blockIdx.y;
  int p = blockIdx.x * 256 + threadIdx.x;
  int y = p >> 8, x = p & 255;
  float acc[8] = {0, 0, 0, 0, 0, 0, 0, 0};
  const float* xb = X + (size_t)b * 64 * HW;
  for (int ic = 0; ic < 64; ++ic) {
    for (int dy = -1; dy <= 1; ++dy)
      for (int dx = -1; dx <= 1; ++dx) {
        int yy = y + dy, xx = x + dx;
        if (yy < 0 || yy > 255 || xx < 0 || xx > 255) continue;
        float v = xb[(unsigned)(ic * HW + yy * 256 + xx)];
        int wi = (dy + 1) * 3 + (dx + 1);
#pragma unroll
        for (int o = 0; o < 8; ++o)
          acc[o] += w[((size_t)(oq * 8 + o) * 64 + ic) * 9 + wi] * v;
      }
  }
  float* ob = out + (size_t)b * 32 * HW;
#pragma unroll
  for (int o = 0; o < 8; ++o) {
    float v = acc[o];
    ob[(unsigned)((oq * 8 + o) * HW + p)] = v > 0.f ? v : 0.01f * v;
  }
}

// head conv 3x3 32->1 + inp_img + sigmoid
__global__ void k_head2(const float* __restrict__ X, const float* __restrict__ w,
                        const float* __restrict__ img, float* __restrict__ out)
{
  int b = blockIdx.y;
  int p = blockIdx.x * 256 + threadIdx.x;
  int y = p >> 8, x = p & 255;
  float acc = 0.f;
  const float* xb = X + (size_t)b * 32 * HW;
  for (int ic = 0; ic < 32; ++ic)
    for (int dy = -1; dy <= 1; ++dy)
      for (int dx = -1; dx <= 1; ++dx) {
        int yy = y + dy, xx = x + dx;
        if (yy < 0 || yy > 255 || xx < 0 || xx > 255) continue;
        acc += w[ic * 9 + (dy + 1) * 3 + (dx + 1)] * xb[(unsigned)(ic * HW + yy * 256 + xx)];
      }
  out[(size_t)b * HW + (unsigned)p] = sigmoidf_(acc + img[(size_t)b * HW + (unsigned)p]);
}

// ---------------------------------------------------------------------------
extern "C" void kernel_launch(void* const* d_in, const int* in_sizes, int n_in,
                              void* d_out, int out_size, void* d_ws, size_t ws_size,
                              hipStream_t stream)
{
  const float* inp_img   = (const float*)d_in[0];
  const float* base_feat = (const float*)d_in[1];
  const float* det_feat  = (const float*)d_in[2];
  const float* w_ir      = (const float*)d_in[3];
  const float* w_vi      = (const float*)d_in[4];
  const float* w_fu      = (const float*)d_in[5];
  const float* b_fu      = (const float*)d_in[6];
  const float* w_dw_ciim = (const float*)d_in[7];
  const float* b_dw_ciim = (const float*)d_in[8];
  const float* eca_w     = (const float*)d_in[9];
  const float* w_rc2     = (const float*)d_in[10];
  const float* ln1_w     = (const float*)d_in[11];
  const float* ln1_b     = (const float*)d_in[12];
  const float* w_qkv     = (const float*)d_in[13];
  const float* w_qkv_dw  = (const float*)d_in[14];
  const float* temp      = (const float*)d_in[15];
  const float* w_proj    = (const float*)d_in[16];
  const float* ln2_w     = (const float*)d_in[17];
  const float* ln2_b     = (const float*)d_in[18];
  const float* w_in      = (const float*)d_in[19];
  const float* w_ffn_dw  = (const float*)d_in[20];
  const float* w_out_ffn = (const float*)d_in[21];
  const float* w_head1   = (const float*)d_in[22];
  const float* w_head2   = (const float*)d_in[23];

  float* out_img = (float*)d_out;                 // [4,1,256,256]
  float* out_x0  = (float*)d_out + 262144;        // [4,64,256,256]

  // workspace layout (floats)
  float* ws    = (float*)d_ws;
  float* bufA  = ws;                               // 256 planes: 4*256*HW
  float* bufC  = bufA + (size_t)4 * 256 * HW;      // 128 planes
  float* bufX  = bufC + (size_t)4 * 128 * HW;      // 64 planes (x)
  float* bufY  = bufX + (size_t)4 * 64 * HW;       // 64 planes (ln / head1)
  float* smallw   = bufY + (size_t)4 * 64 * HW;
  float* meansLH  = smallw;            // 4*256
  float* outMeans = smallw + 1024;     // 4*128
  float* gatev    = smallw + 1536;     // 4*128
  float* gram     = smallw + 2048;     // 32*80
  float* attn     = smallw + 4608;     // 32*64
  _Float16* hW    = (_Float16*)(smallw + 8192);
  _Float16* h_ir   = hW;
  _Float16* h_vi   = hW + 8192;
  _Float16* h_fu   = hW + 16384;
  _Float16* h_rc2  = hW + 49152;
  _Float16* h_qkv  = hW + 57344;   // 4 * 192*64
  _Float16* h_proj = hW + 106496;  // 4 * 64*64
  _Float16* h_in   = hW + 122880;  // 4 * 256*64
  _Float16* h_out  = hW + 188416;  // 4 * 64*128

  auto cvt = [stream](const float* s, _Float16* d, int n) {
    k_cvt<<<dim3((n + 255) / 256), dim3(256), 0, stream>>>(s, d, n);
  };
  cvt(w_ir, h_ir, 8192);
  cvt(w_vi, h_vi, 8192);
  cvt(w_fu, h_fu, 32768);
  cvt(w_rc2, h_rc2, 8192);
  cvt(w_qkv, h_qkv, 4 * 12288);
  cvt(w_proj, h_proj, 4 * 4096);
  cvt(w_in, h_in, 4 * 16384);
  cvt(w_out_ffn, h_out, 4 * 8192);

  const dim3 gG(HW / 32, BATCH), bG(256);   // GEMM grid
  const dim3 gP(HW / 256, BATCH), bP(256);  // per-pixel grid

  // ---- CIIM ----
  // lf / hf into concat layout of bufA (256 planes)
  gemm1x1<64, false, false, false, false><<<gG, bG, 0, stream>>>(
      base_feat, 64, 0, h_ir, nullptr, nullptr, nullptr, bufA, nullptr, 128, 256, 0);
  gemm1x1<64, false, false, false, false><<<gG, bG, 0, stream>>>(
      det_feat, 64, 0, h_vi, nullptr, nullptr, nullptr, bufA, nullptr, 128, 256, 128);
  k_chan_mean<<<dim3(256, BATCH), dim3(256), 0, stream>>>(bufA, meansLH, 256);
  // fuse = W_fu * concat(lf,hf) + b
  gemm1x1<256, true, false, false, false><<<gG, bG, 0, stream>>>(
      bufA, 256, 0, h_fu, b_fu, nullptr, nullptr, bufC, nullptr, 128, 128, 0);
  // grouped depthwise + gates + shuffle -> bufA [b][128][HW]
  k_ciim_dw<<<dim3(HW / 256, 64, BATCH), dim3(256), 0, stream>>>(
      bufC, meansLH, base_feat, det_feat, w_dw_ciim, b_dw_ciim, bufA);
  k_chan_mean<<<dim3(128, BATCH), dim3(256), 0, stream>>>(bufA, outMeans, 128);
  k_eca<<<dim3(2), dim3(256), 0, stream>>>(outMeans, eca_w, gatev);
  // x0 = W_rc2 * (gate .* out)  -> d_out x0 region AND bufX
  gemm1x1<128, false, false, true, true><<<gG, bG, 0, stream>>>(
      bufA, 128, 0, h_rc2, nullptr, gatev, nullptr, bufX, out_x0, 64, 64, 0);

  // ---- transformer blocks ----
  for (int i = 0; i < 4; ++i) {
    // attention
    k_ln<<<gP, bP, 0, stream>>>(bufX, ln1_w + i * 64, ln1_b + i * 64, bufY);
    gemm1x1<64, false, false, false, false><<<gG, bG, 0, stream>>>(
        bufY, 64, 0, h_qkv + i * 12288, nullptr, nullptr, nullptr, bufA, nullptr, 192, 192, 0);
    k_zero<<<dim3(10), dim3(256), 0, stream>>>(gram, 2560);
    k_gram<<<dim3(64, 8, BATCH), dim3(256), 0, stream>>>(bufA, w_qkv_dw + i * 192 * 9, gram);
    k_softmax<<<dim3(8, BATCH), dim3(64), 0, stream>>>(gram, temp + i * 8, attn);
    k_attn_apply<<<gP, bP, 0, stream>>>(bufA, w_qkv_dw + i * 192 * 9, attn, bufC);
    gemm1x1<64, false, true, false, false><<<gG, bG, 0, stream>>>(
        bufC, 64, 0, h_proj + i * 4096, nullptr, nullptr, bufX, bufX, nullptr, 64, 64, 0);
    // ffn
    k_ln<<<gP, bP, 0, stream>>>(bufX, ln2_w + i * 64, ln2_b + i * 64, bufY);
    gemm1x1<64, false, false, false, false><<<gG, bG, 0, stream>>>(
        bufY, 64, 0, h_in + i * 16384, nullptr, nullptr, nullptr, bufA, nullptr, 256, 256, 0);
    k_ffn_gate<<<gP, bP, 0, stream>>>(bufA, w_ffn_dw + i * 256 * 9, bufC);
    gemm1x1<128, false, true, false, false><<<gG, bG, 0, stream>>>(
        bufC, 128, 0, h_out + i * 8192, nullptr, nullptr, bufX, bufX, nullptr, 64, 64, 0);
  }

  // ---- head ----
  k_head1<<<dim3(HW / 256, 4, BATCH), dim3(256), 0, stream>>>(bufX, w_head1, bufY);
  k_head2<<<gP, bP, 0, stream>>>(bufY, w_head2, inp_img, out_img);
}
